// DecoderRNN_75522704933094
// MI455X (gfx1250) — matrix operand order repaired and checked by reading, not compiled, not run on hardware
//
#include <hip/hip_runtime.h>
#include <hip/hip_bf16.h>
#include <cstdint>

typedef __attribute__((ext_vector_type(16))) __bf16 v16bf;
typedef __attribute__((ext_vector_type(8)))  float  v8f;

#define B_     64
#define P_     49
#define ENC_   1280
#define TSTEPS 24
#define V_     10000
#define E_     512
#define H_     1024
#define A_     512

__device__ __forceinline__ float sigmoidf_(float x) { return 1.f / (1.f + __expf(-x)); }

// ---------------------------------------------------------------------------
// Pack weights into bf16 WMMA B-operand tiles.
// Logical weight matrix Wlog[Ntot, Ktot]:
//   rows n <  NA : K-concat  Wlog[n,k] = (k<K1) ? W1[n,k] : W2[n,k-K1]
//   rows n >= NA : region B  Wlog[n,k] = WB[(n-NA)*Ktot + k]
//   n >= Ntot    : zero (padding for NT-aligned tile groups)
// Tile layout per (kt, ntile): 32 lanes x 16 bf16, lane%16 = n col,
// lane/16 = K half-group, element e = K offset within half-group.
// ---------------------------------------------------------------------------
__global__ void pack_w_kernel(const float* __restrict__ W1, int K1,
                              const float* __restrict__ W2, int K2,
                              const float* __restrict__ WB, int NA, int Ntot,
                              __bf16* __restrict__ out, int ntilesPad)
{
    const int  Ktot = K1 + K2;
    const int  ktt  = Ktot >> 5;
    long idx = (long)blockIdx.x * blockDim.x + threadIdx.x;
    long total = (long)ktt * ntilesPad * 32;
    if (idx >= total) return;
    const int  lane = (int)(idx & 31);
    const long tile = idx >> 5;
    const int  nt   = (int)(tile % ntilesPad);
    const int  kt   = (int)(tile / ntilesPad);
    const int  n    = nt * 16 + (lane & 15);
    const int  kb   = kt * 32 + ((lane >> 4) << 4);

    __align__(16) __bf16 vals[16];
#pragma unroll
    for (int e = 0; e < 16; ++e) {
        int   k = kb + e;
        float v = 0.f;
        if (n < NA)        v = (k < K1) ? W1[(size_t)n * K1 + k] : W2[(size_t)n * K2 + (k - K1)];
        else if (n < Ntot) v = WB[(size_t)(n - NA) * Ktot + k];
        vals[e] = (__bf16)v;
    }
    uint32_t* o = (uint32_t*)(out + ((size_t)tile * 32 + lane) * 16);
    const uint32_t* s = (const uint32_t*)vals;
#pragma unroll
    for (int i = 0; i < 8; ++i) o[i] = s[i];
}

// ---------------------------------------------------------------------------
// C = [X1 | X2] @ Wlog^T + bias, f32 accumulate via v_wmma_f32_16x16x32_bf16.
// One wave per (16-row tile, NT 16-col tiles). grid = (M/16, ntilesPad/NT).
// Dual-region output: logical cols [0,N1) -> C (stride ldc, bias biasA1+biasA2),
// cols [N1,N1+N2) -> C2 (stride ldc2, bias biasB). All NT B tiles are loaded
// into a register array before the WMMA chain so 2*NT b128 loads are in flight.
// ---------------------------------------------------------------------------
template <int NT>
__global__ void wmma_gemm_kernel(const float* __restrict__ X1, int K1,
                                 const float* __restrict__ X2, int K2,
                                 const __bf16* __restrict__ Wp, int ntilesPad,
                                 const float* __restrict__ biasA1,
                                 const float* __restrict__ biasA2,
                                 float* __restrict__ C, int N1, int ldc,
                                 const float* __restrict__ biasB,
                                 float* __restrict__ C2, int N2, int ldc2)
{
    const int lane = threadIdx.x;                 // wave32
    const int mt   = blockIdx.x;
    const int n0t  = blockIdx.y * NT;
    const int Ktot = K1 + K2;
    const int ktt  = Ktot >> 5;
    const int ncol = lane & 15;
    const int koff = (lane >> 4) << 4;            // A K-contiguous: lanes16-31 start at +16
    const int row  = mt * 16 + ncol;

    v8f acc[NT];
#pragma unroll
    for (int j = 0; j < NT; ++j) {
        int   n  = (n0t + j) * 16 + ncol;
        float bv = 0.f;
        if (n < N1) {
            if (biasA1) bv += biasA1[n];
            if (biasA2) bv += biasA2[n];
        } else if (n < N1 + N2) {
            if (biasB) bv += biasB[n - N1];
        }
        acc[j] = (v8f){bv, bv, bv, bv, bv, bv, bv, bv};
    }

    for (int kt = 0; kt < ktt; ++kt) {
        const int kg = kt << 5;
        const float* xp;
        if (kg < K1) xp = X1 + (size_t)row * K1 + kg;
        else         xp = X2 + (size_t)row * K2 + (kg - K1);

        float4 f0 = *(const float4*)(xp + koff);
        float4 f1 = *(const float4*)(xp + koff + 4);
        float4 f2 = *(const float4*)(xp + koff + 8);
        float4 f3 = *(const float4*)(xp + koff + 12);

        // cluster all B-tile loads so 2*NT b128 loads are outstanding
        union { float4 f4[2]; v16bf v; } bu[NT];
        const __bf16* bbase = Wp + (((size_t)kt * ntilesPad + n0t) * 32 + lane) * 16;
#pragma unroll
        for (int j = 0; j < NT; ++j) {
            const float4* bp = (const float4*)(bbase + (size_t)j * 512);
            bu[j].f4[0] = bp[0];
            bu[j].f4[1] = bp[1];
        }

        v16bf a;
        a[0]  = (__bf16)f0.x; a[1]  = (__bf16)f0.y; a[2]  = (__bf16)f0.z; a[3]  = (__bf16)f0.w;
        a[4]  = (__bf16)f1.x; a[5]  = (__bf16)f1.y; a[6]  = (__bf16)f1.z; a[7]  = (__bf16)f1.w;
        a[8]  = (__bf16)f2.x; a[9]  = (__bf16)f2.y; a[10] = (__bf16)f2.z; a[11] = (__bf16)f2.w;
        a[12] = (__bf16)f3.x; a[13] = (__bf16)f3.y; a[14] = (__bf16)f3.z; a[15] = (__bf16)f3.w;

#pragma unroll
        for (int j = 0; j < NT; ++j)
            acc[j] = __builtin_amdgcn_wmma_f32_16x16x32_bf16(
                false, a, false, bu[j].v, (short)0, acc[j], false, false);
    }

    const int rbase = mt * 16 + ((lane >> 4) << 3);
#pragma unroll
    for (int j = 0; j < NT; ++j) {
        int n = (n0t + j) * 16 + ncol;
        if (n < N1) {
#pragma unroll
            for (int r = 0; r < 8; ++r)
                C[(size_t)(rbase + r) * ldc + n] = acc[j][r];
        } else if (n < N1 + N2) {
#pragma unroll
            for (int r = 0; r < 8; ++r)
                C2[(size_t)(rbase + r) * ldc2 + (n - N1)] = acc[j][r];
        }
    }
}

// ---------------------------------------------------------------------------
__global__ void mean_enc_kernel(const float* __restrict__ enc, float* __restrict__ out)
{
    int idx = blockIdx.x * blockDim.x + threadIdx.x;
    if (idx >= B_ * ENC_) return;
    int b = idx / ENC_, c = idx % ENC_;
    float s = 0.f;
    for (int p = 0; p < P_; ++p) s += enc[((size_t)b * P_ + p) * ENC_ + c];
    out[idx] = s * (1.f / P_);
}

// embT[b,t,e] = embed[captions[b,t], e]  (t = 0..TSTEPS-1, captions has T=25)
__global__ void gather_emb_kernel(const int* __restrict__ cap,
                                  const float* __restrict__ embed,
                                  float* __restrict__ out)
{
    int idx = blockIdx.x * blockDim.x + threadIdx.x;
    if (idx >= B_ * TSTEPS * E_) return;
    int e = idx % E_, bt = idx / E_;
    int t = bt % TSTEPS, b = bt / TSTEPS;
    int tok = cap[b * 25 + t];
    out[idx] = embed[(size_t)tok * E_ + e];
}

// ---------------------------------------------------------------------------
// Per-step attention: e_p = relu(att1+att2)@Wf + bf; softmax over P;
// ctx = alpha @ enc. Writes [x_t | ctx] into gin[b, 0:1792]. 1 block / batch.
// ---------------------------------------------------------------------------
__global__ void attention_kernel(const float* __restrict__ att1,
                                 const float* __restrict__ att2,
                                 const float* __restrict__ Wf,
                                 const float* __restrict__ bf,
                                 const float* __restrict__ enc,
                                 const float* __restrict__ embT, int t,
                                 float* __restrict__ gin)
{
    const int b = blockIdx.x;
    const int tid = threadIdx.x;
    const int wave = tid >> 5, lane = tid & 31;
    __shared__ float s_e[P_];

    const float* a2 = att2 + (size_t)b * A_;
    for (int p = wave; p < P_; p += 8) {
        const float* a1 = att1 + ((size_t)b * P_ + p) * A_;
        float s = 0.f;
        for (int a = lane; a < A_; a += 32) {
            float v = a1[a] + a2[a];
            v = v > 0.f ? v : 0.f;
            s += v * Wf[a];
        }
        for (int off = 16; off > 0; off >>= 1) s += __shfl_down(s, off);
        if (lane == 0) s_e[p] = s + bf[0];
    }
    __syncthreads();

    float mx = -1e30f;
    for (int p = 0; p < P_; ++p) mx = fmaxf(mx, s_e[p]);
    __syncthreads();
    if (tid < P_) s_e[tid] = __expf(s_e[tid] - mx);
    __syncthreads();
    float den = 0.f;
    for (int p = 0; p < P_; ++p) den += s_e[p];
    const float inv = 1.f / den;

    for (int cc = tid; cc < E_; cc += blockDim.x)
        gin[(size_t)b * (E_ + ENC_) + cc] = embT[((size_t)b * TSTEPS + t) * E_ + cc];
    for (int cc = tid; cc < ENC_; cc += blockDim.x) {
        float s = 0.f;
        for (int p = 0; p < P_; ++p)
            s += s_e[p] * enc[((size_t)b * P_ + p) * ENC_ + cc];
        gin[(size_t)b * (E_ + ENC_) + E_ + cc] = s * inv;
    }
}

// LSTM cell: gates[b, 4H] -> updates h, c in place
__global__ void lstm_cell_kernel(const float* __restrict__ gates,
                                 float* __restrict__ h, float* __restrict__ c)
{
    int idx = blockIdx.x * blockDim.x + threadIdx.x;
    if (idx >= B_ * H_) return;
    int b = idx >> 10, n = idx & (H_ - 1);
    const float* g = gates + (size_t)b * 4 * H_;
    float ig = sigmoidf_(g[n]);
    float fg = sigmoidf_(g[n + H_]);
    float gg = tanhf(g[n + 2 * H_]);
    float og = sigmoidf_(g[n + 3 * H_]);
    float cn = fg * c[idx] + ig * gg;
    c[idx] = cn;
    h[idx] = og * tanhf(cn);
}

// ---------------------------------------------------------------------------
extern "C" void kernel_launch(void* const* d_in, const int* in_sizes, int n_in,
                              void* d_out, int out_size, void* d_ws, size_t ws_size,
                              hipStream_t stream)
{
    const float* enc    = (const float*)d_in[0];
    const int*   cap    = (const int*)d_in[1];
    const float* embed  = (const float*)d_in[2];
    const float* W_eatt = (const float*)d_in[3];
    const float* b_eatt = (const float*)d_in[4];
    const float* W_datt = (const float*)d_in[5];
    const float* b_datt = (const float*)d_in[6];
    const float* W_fatt = (const float*)d_in[7];
    const float* b_fatt = (const float*)d_in[8];
    const float* W_ih   = (const float*)d_in[9];
    const float* b_ih   = (const float*)d_in[10];
    const float* W_hh   = (const float*)d_in[11];
    const float* b_hh   = (const float*)d_in[12];
    const float* W_h0   = (const float*)d_in[13];
    const float* b_h0   = (const float*)d_in[14];
    const float* W_c0   = (const float*)d_in[15];
    const float* b_c0   = (const float*)d_in[16];
    const float* W_out  = (const float*)d_in[17];
    const float* b_out  = (const float*)d_in[18];
    float* out = (float*)d_out;

    char* ws = (char*)d_ws;
    size_t off = 0;
    auto alloc = [&](size_t bytes) -> char* {
        char* p = ws + off;
        off += (bytes + 255) & ~(size_t)255;
        return p;
    };

    // padded N-tile counts (multiples of NT=8)
    const int ntWcomb = 256;   // gates: N=4096 -> 256 tiles
    const int ntWod   = 664;   // fused [W_out;W_datt]: N=10512 -> 657 tiles -> pad 664
    const int ntWdatt = 32;    // 512/16 (used once for att2_0)
    const int ntWeatt = 32;
    const int ntWh0   = 64;

    __bf16* pWcomb = (__bf16*)alloc((size_t)88 * ntWcomb * 512 * 2);  // K=2816
    __bf16* pWod   = (__bf16*)alloc((size_t)32 * ntWod   * 512 * 2);  // K=1024
    __bf16* pWdatt = (__bf16*)alloc((size_t)32 * ntWdatt * 512 * 2);  // K=1024
    __bf16* pWeatt = (__bf16*)alloc((size_t)40 * ntWeatt * 512 * 2);  // K=1280
    __bf16* pWh0   = (__bf16*)alloc((size_t)40 * ntWh0   * 512 * 2);  // K=1280
    __bf16* pWc0   = (__bf16*)alloc((size_t)40 * ntWh0   * 512 * 2);
    float* att1  = (float*)alloc((size_t)B_ * P_ * A_ * 4);
    float* embT  = (float*)alloc((size_t)B_ * TSTEPS * E_ * 4);
    float* meanb = (float*)alloc((size_t)B_ * ENC_ * 4);
    float* hbuf  = (float*)alloc((size_t)B_ * H_ * 4);
    float* cbuf  = (float*)alloc((size_t)B_ * H_ * 4);
    float* att2  = (float*)alloc((size_t)B_ * A_ * 4);
    float* gin   = (float*)alloc((size_t)B_ * (E_ + ENC_) * 4);
    float* gates = (float*)alloc((size_t)B_ * 4 * H_ * 4);

    auto cdiv = [](long a, long b) { return (int)((a + b - 1) / b); };

    // --- one-time prep (per launch) ---
    pack_w_kernel<<<cdiv(88L * ntWcomb * 32, 256), 256, 0, stream>>>(
        W_ih, 1792, W_hh, 1024, nullptr, 4096, 4096, pWcomb, ntWcomb);
    pack_w_kernel<<<cdiv(32L * ntWod * 32, 256), 256, 0, stream>>>(
        W_out, 1024, nullptr, 0, W_datt, V_, V_ + A_, pWod, ntWod);
    pack_w_kernel<<<cdiv(32L * ntWdatt * 32, 256), 256, 0, stream>>>(
        W_datt, 1024, nullptr, 0, nullptr, A_, A_, pWdatt, ntWdatt);
    pack_w_kernel<<<cdiv(40L * ntWeatt * 32, 256), 256, 0, stream>>>(
        W_eatt, 1280, nullptr, 0, nullptr, A_, A_, pWeatt, ntWeatt);
    pack_w_kernel<<<cdiv(40L * ntWh0 * 32, 256), 256, 0, stream>>>(
        W_h0, 1280, nullptr, 0, nullptr, H_, H_, pWh0, ntWh0);
    pack_w_kernel<<<cdiv(40L * ntWh0 * 32, 256), 256, 0, stream>>>(
        W_c0, 1280, nullptr, 0, nullptr, H_, H_, pWc0, ntWh0);

    mean_enc_kernel<<<cdiv(B_ * ENC_, 256), 256, 0, stream>>>(enc, meanb);
    gather_emb_kernel<<<cdiv((long)B_ * TSTEPS * E_, 256), 256, 0, stream>>>(cap, embed, embT);

    // h0 / c0 = mean_enc @ W^T + b
    wmma_gemm_kernel<8><<<dim3(B_ / 16, ntWh0 / 8), 32, 0, stream>>>(
        meanb, 1280, nullptr, 0, pWh0, ntWh0, b_h0, nullptr, hbuf, H_, H_,
        nullptr, nullptr, 0, 0);
    wmma_gemm_kernel<8><<<dim3(B_ / 16, ntWh0 / 8), 32, 0, stream>>>(
        meanb, 1280, nullptr, 0, pWc0, ntWh0, b_c0, nullptr, cbuf, H_, H_,
        nullptr, nullptr, 0, 0);
    // att1 = enc @ W_eatt^T + b_eatt  (M = B*P = 3136)
    wmma_gemm_kernel<8><<<dim3((B_ * P_) / 16, ntWeatt / 8), 32, 0, stream>>>(
        enc, 1280, nullptr, 0, pWeatt, ntWeatt, b_eatt, nullptr, att1, A_, A_,
        nullptr, nullptr, 0, 0);
    // att2_0 = h0 @ W_datt^T + b_datt (subsequent att2 fused into logits GEMM)
    wmma_gemm_kernel<8><<<dim3(B_ / 16, ntWdatt / 8), 32, 0, stream>>>(
        hbuf, 1024, nullptr, 0, pWdatt, ntWdatt, b_datt, nullptr, att2, A_, A_,
        nullptr, nullptr, 0, 0);

    // --- sequential decode loop: 4 kernels per step ---
    for (int t = 0; t < TSTEPS; ++t) {
        // softmax attention + context + x_t staging (att2 from previous fused GEMM)
        attention_kernel<<<B_, 256, 0, stream>>>(att1, att2, W_fatt, b_fatt, enc, embT, t, gin);
        // gates = [x_t|ctx] @ W_ih^T + h @ W_hh^T + b_ih + b_hh  (fused K = 2816)
        wmma_gemm_kernel<8><<<dim3(B_ / 16, ntWcomb / 8), 32, 0, stream>>>(
            gin, 1792, hbuf, 1024, pWcomb, ntWcomb, b_ih, b_hh, gates, 4 * H_, 4 * H_,
            nullptr, nullptr, 0, 0);
        lstm_cell_kernel<<<cdiv(B_ * H_, 256), 256, 0, stream>>>(gates, hbuf, cbuf);
        // fused: preds_t = h_new @ W_out^T + b_out (strided into [B,24,V])
        //        att2_{t+1} = h_new @ W_datt^T + b_datt
        wmma_gemm_kernel<8><<<dim3(B_ / 16, ntWod / 8), 32, 0, stream>>>(
            hbuf, 1024, nullptr, 0, pWod, ntWod, b_out, nullptr,
            out + (size_t)t * V_, V_, TSTEPS * V_,
            b_datt, att2, A_, A_);
    }

    (void)in_sizes; (void)n_in; (void)out_size; (void)ws_size;
}
